// Model_33921651704491
// MI455X (gfx1250) — matrix operand order repaired and checked by reading, not compiled, not run on hardware
//
#include <hip/hip_runtime.h>
#include <hip/hip_bf16.h>
#include <math.h>

// Problem constants (match reference)
#define BATCH 64
#define SEQ   512
#define HID   1024
#define KTOT  2048   // concat(x_t, h) reduction dim
#define GATES 4096   // 4*HID

typedef __attribute__((ext_vector_type(16))) __bf16 v16bf;
typedef __attribute__((ext_vector_type(8)))  __bf16 v8bf;
typedef __attribute__((ext_vector_type(8)))  float  v8f;

// ---------------------------------------------------------------------------
// Setup 1: pack W = [w_ih | w_hh] into one row-major bf16 matrix [4096 x 2048]
// ---------------------------------------------------------------------------
__global__ void pack_weights(const float* __restrict__ w_ih,
                             const float* __restrict__ w_hh,
                             __bf16* __restrict__ wcat) {
    size_t i = (size_t)blockIdx.x * blockDim.x + threadIdx.x;
    if (i >= (size_t)GATES * KTOT) return;
    int n = (int)(i >> 11);          // /2048
    int k = (int)(i & 2047);
    float v = (k < HID) ? w_ih[(size_t)n * HID + k]
                        : w_hh[(size_t)n * HID + (k - HID)];
    wcat[i] = (__bf16)v;
}

// ---------------------------------------------------------------------------
// Setup 2: convert the whole input x to bf16 once (read-once from HBM)
// ---------------------------------------------------------------------------
__global__ void convert_x(const float* __restrict__ x, __bf16* __restrict__ xb,
                          size_t n) {
    size_t i = (size_t)blockIdx.x * blockDim.x + threadIdx.x;
    if (i < n) xb[i] = (__bf16)x[i];
}

// ---------------------------------------------------------------------------
// Setup 3: bias = b_ih + b_hh ; h(bf16) <- h0 ; c <- c0
// ---------------------------------------------------------------------------
__global__ void init_state(const float* __restrict__ h0,
                           const float* __restrict__ c0,
                           const float* __restrict__ b_ih,
                           const float* __restrict__ b_hh,
                           __bf16* __restrict__ hb,
                           float*  __restrict__ c,
                           float*  __restrict__ bias) {
    int i = blockIdx.x * blockDim.x + threadIdx.x;   // 0 .. 65535
    if (i < GATES) bias[i] = b_ih[i] + b_hh[i];
    if (i < BATCH * HID) { hb[i] = (__bf16)h0[i]; c[i] = c0[i]; }
}

__device__ __forceinline__ float sigmoidf_(float x) {
    return 1.0f / (1.0f + __expf(-x));
}

// ---------------------------------------------------------------------------
// One LSTM timestep. Grid: 64 blocks (nTile), 256 threads = 8 waves.
// wave id = (kHalf, mTile): kHalf=0 -> x-GEMM (K 0..1023, reads bf16 x_t),
//                           kHalf=1 -> h-GEMM (K 1024..2047, reads bf16 h).
// Each wave: 4 gate accumulators (i,f,g,o) for a 16x16 (batch x hidden) tile,
// 128 WMMAs, all-bf16 operands (no in-loop conversion). The two K-halves are
// combined via a 16 KB LDS reduction; kHalf=0 waves apply the fused
// sigmoid/tanh cell update in registers and write c (f32) + h (bf16).
// ---------------------------------------------------------------------------
__global__ __launch_bounds__(256) void lstm_step(
    const __bf16* __restrict__ xb,     // [B, S, HID] bf16
    const __bf16* __restrict__ wcat,   // [4096, 2048] bf16 row-major
    const float*  __restrict__ bias,   // [4096]
    const __bf16* __restrict__ hb_in,  // [B, HID] bf16
    __bf16*       __restrict__ hb_out, // [B, HID] bf16
    float*        __restrict__ c,      // [B, HID] f32
    float*        __restrict__ h_f32,  // [B, HID] f32 (written only if last)
    int t, int last)
{
    __shared__ float red[4][4][32][8];             // [mTile][gate][lane][r]

    const int lane  = threadIdx.x & 31;
    const int wid   = threadIdx.x >> 5;            // 0..7
    const int mTile = wid & 3;
    const int kHalf = wid >> 2;
    const int nTile = blockIdx.x;                  // 0..63
    const int mBase = mTile * 16;
    const int nBase = nTile * 16;

    // A-fragment lane geometry (16x32 bf16):
    //   lane<16 : M=lane,    K {kb+0..15}
    //   lane>=16: M=lane-16, K {kb+16..31}
    const int aM    = lane & 15;
    const int aKoff = (lane >> 4) * 16;
    // B-fragment lane geometry (32x16): lane = column, 16 contiguous K
    const int bN    = lane & 15;
    const int bKoff = (lane >> 4) * 16;

    const __bf16* arow = kHalf
        ? (hb_in + (size_t)(mBase + aM) * HID)
        : (xb + (size_t)(mBase + aM) * (SEQ * HID) + (size_t)t * HID);

    const size_t wko = (size_t)(kHalf ? HID : 0) + (size_t)bKoff;
    const __bf16* w0 = wcat + ((size_t)(0 * HID + nBase + bN)) * KTOT + wko;
    const __bf16* w1 = wcat + ((size_t)(1 * HID + nBase + bN)) * KTOT + wko;
    const __bf16* w2 = wcat + ((size_t)(2 * HID + nBase + bN)) * KTOT + wko;
    const __bf16* w3 = wcat + ((size_t)(3 * HID + nBase + bN)) * KTOT + wko;

    v8f acc0 = {}, acc1 = {}, acc2 = {}, acc3 = {};

#pragma unroll 4
    for (int kb = 0; kb < HID; kb += 32) {
        v8bf lo = *(const v8bf*)(arow + kb + aKoff);
        v8bf hi = *(const v8bf*)(arow + kb + 8 + aKoff);
        v16bf a;
#pragma unroll
        for (int i = 0; i < 8; ++i) { a[i] = lo[i]; a[8 + i] = hi[i]; }
        v16bf b0 = *(const v16bf*)(w0 + kb);
        v16bf b1 = *(const v16bf*)(w1 + kb);
        v16bf b2 = *(const v16bf*)(w2 + kb);
        v16bf b3 = *(const v16bf*)(w3 + kb);
        acc0 = __builtin_amdgcn_wmma_f32_16x16x32_bf16(false, a, false, b0, (short)0, acc0, false, false);
        acc1 = __builtin_amdgcn_wmma_f32_16x16x32_bf16(false, a, false, b1, (short)0, acc1, false, false);
        acc2 = __builtin_amdgcn_wmma_f32_16x16x32_bf16(false, a, false, b2, (short)0, acc2, false, false);
        acc3 = __builtin_amdgcn_wmma_f32_16x16x32_bf16(false, a, false, b3, (short)0, acc3, false, false);
    }

    // ---- combine the two K-halves through LDS ----
    if (kHalf == 1) {
#pragma unroll
        for (int r = 0; r < 8; ++r) {
            red[mTile][0][lane][r] = acc0[r];
            red[mTile][1][lane][r] = acc1[r];
            red[mTile][2][lane][r] = acc2[r];
            red[mTile][3][lane][r] = acc3[r];
        }
    }
    __syncthreads();
    if (kHalf == 0) {
        // D layout: lane holds column N = nBase+(lane&15); vgpr r holds
        // row M = mBase + 8*(lane>>4) + r.
        const int nIdx = nBase + (lane & 15);
        const int mOff = mBase + (lane >> 4) * 8;
        const float bi = bias[0 * HID + nIdx];
        const float bf = bias[1 * HID + nIdx];
        const float bg = bias[2 * HID + nIdx];
        const float bo = bias[3 * HID + nIdx];

#pragma unroll
        for (int r = 0; r < 8; ++r) {
            const size_t idx = (size_t)(mOff + r) * HID + nIdx;
            float iv = sigmoidf_(acc0[r] + red[mTile][0][lane][r] + bi);
            float fv = sigmoidf_(acc1[r] + red[mTile][1][lane][r] + bf);
            float gv = tanhf(acc2[r] + red[mTile][2][lane][r] + bg);
            float ov = sigmoidf_(acc3[r] + red[mTile][3][lane][r] + bo);
            float cv = fv * c[idx] + iv * gv;
            c[idx] = cv;
            float hv = ov * tanhf(cv);
            hb_out[idx] = (__bf16)hv;
            if (last) h_f32[idx] = hv;
        }
    }
}

// ---------------------------------------------------------------------------
// Host-side launch
// ---------------------------------------------------------------------------
extern "C" void kernel_launch(void* const* d_in, const int* in_sizes, int n_in,
                              void* d_out, int out_size, void* d_ws, size_t ws_size,
                              hipStream_t stream) {
    (void)in_sizes; (void)n_in; (void)out_size; (void)ws_size;

    const float* x    = (const float*)d_in[0];  // [64,512,1024]
    const float* h0   = (const float*)d_in[1];  // [1,64,1024]
    const float* c0   = (const float*)d_in[2];  // [1,64,1024]
    const float* w_ih = (const float*)d_in[3];  // [4096,1024]
    const float* w_hh = (const float*)d_in[4];  // [4096,1024]
    const float* b_ih = (const float*)d_in[5];  // [4096]
    const float* b_hh = (const float*)d_in[6];  // [4096]

    // Workspace carve-out (~84 MB total)
    char* ws = (char*)d_ws;
    __bf16* wcat = (__bf16*)ws;                                  // 16 MB
    ws += (size_t)GATES * KTOT * sizeof(__bf16);
    __bf16* xb   = (__bf16*)ws;                                  // 67 MB
    ws += (size_t)BATCH * SEQ * HID * sizeof(__bf16);
    float* bias  = (float*)ws;  ws += GATES * sizeof(float);     // 16 KB
    float* hf32  = (float*)ws;  ws += BATCH * HID * sizeof(float);
    float* cbuf  = (float*)ws;  ws += BATCH * HID * sizeof(float);
    __bf16* hb0  = (__bf16*)ws; ws += (size_t)BATCH * HID * sizeof(__bf16);
    __bf16* hb1  = (__bf16*)ws; ws += (size_t)BATCH * HID * sizeof(__bf16);

    // Pack weights to bf16 (L2-resident for the whole recurrence)
    {
        size_t n = (size_t)GATES * KTOT;
        pack_weights<<<(int)((n + 255) / 256), 256, 0, stream>>>(w_ih, w_hh, wcat);
    }
    // Convert x to bf16 once
    {
        size_t n = (size_t)BATCH * SEQ * HID;
        convert_x<<<(int)((n + 255) / 256), 256, 0, stream>>>(x, xb, n);
    }
    // Init bias / h(bf16) / c
    init_state<<<(BATCH * HID + 255) / 256, 256, 0, stream>>>(
        h0, c0, b_ih, b_hh, hb0, cbuf, bias);

    // 512 sequential steps, double-buffered bf16 h
    for (int t = 0; t < SEQ; ++t) {
        __bf16* hin  = (t & 1) ? hb1 : hb0;
        __bf16* hout = (t & 1) ? hb0 : hb1;
        lstm_step<<<dim3(HID / 16), dim3(256), 0, stream>>>(
            xb, wcat, bias, hin, hout, cbuf, hf32, t, (t == SEQ - 1) ? 1 : 0);
    }
    // Final h (f32) -> output
    hipMemcpyAsync(d_out, hf32, (size_t)BATCH * HID * sizeof(float),
                   hipMemcpyDeviceToDevice, stream);
}
